// TransformerClassifier_6055903888176
// MI455X (gfx1250) — matrix operand order repaired and checked by reading, not compiled, not run on hardware
//
#include <hip/hip_runtime.h>
#include <hip/hip_bf16.h>

// ---------------------------------------------------------------------------
// Types for CDNA5 WMMA (gfx1250, wave32)
// ---------------------------------------------------------------------------
typedef __attribute__((ext_vector_type(16))) __bf16 v16bf;
typedef __attribute__((ext_vector_type(8)))  float  v8f;

__device__ __forceinline__ __bf16 f2bf(float f) { return (__bf16)f; }

__device__ __forceinline__ v8f vzero8() {
    v8f z;
#pragma unroll
    for (int i = 0; i < 8; ++i) z[i] = 0.0f;
    return z;
}

__device__ __forceinline__ float gelu_exact(float x) {
    return 0.5f * x * (1.0f + erff(x * 0.7071067811865476f));
}

// LDS byte-offset of a generic pointer to __shared__ (low 32 bits on amdgcn)
__device__ __forceinline__ unsigned lds_off32(const void* p) {
    return (unsigned)(unsigned long long)p;
}

// Async DMA: 16B per lane, global -> LDS, tracked by ASYNCcnt (no VGPR data).
__device__ __forceinline__ void async_g2l_b128(unsigned loff, const void* gaddr) {
    asm volatile("global_load_async_to_lds_b128 %0, %1, off"
                 :: "v"(loff), "v"(gaddr) : "memory");
}
__device__ __forceinline__ void wait_asynccnt0() {
    asm volatile("s_wait_asynccnt 0x0" ::: "memory");
}
// async loads complete in order: <=2 outstanding => oldest tile has landed
__device__ __forceinline__ void wait_asynccnt2() {
    asm volatile("s_wait_asynccnt 0x2" ::: "memory");
}
__device__ __forceinline__ void wait_dscnt0() {
    asm volatile("s_wait_dscnt 0x0" ::: "memory");
}

// ---------------------------------------------------------------------------
// fp32 -> bf16 weight pre-conversion (run once per launch; L2-resident after)
// ---------------------------------------------------------------------------
__global__ void cvt_w_bf16(const float* __restrict__ src,
                           __bf16* __restrict__ dst, int n)
{
    int i = blockIdx.x * 256 + threadIdx.x;
    if (i < n) dst[i] = (__bf16)src[i];
}

// ---------------------------------------------------------------------------
// Conv frontend: depthwise(k=7, stride 2) -> pointwise -> BN -> GELU
// ---------------------------------------------------------------------------
__global__ __launch_bounds__(64) void conv1_k(
    const float* __restrict__ x, const float* __restrict__ dw,
    const float* __restrict__ pw, const float* __restrict__ g,
    const float* __restrict__ bb, float* __restrict__ out)
{
    const int to = blockIdx.x;   // 0..2047
    const int b  = blockIdx.y;   // 0..7
    const int c  = threadIdx.x;  // 0..63
    __shared__ float sm[64];
    float acc = 0.0f;
    const int t0 = 2 * to - 3;
#pragma unroll
    for (int j = 0; j < 7; ++j) {
        int ti = t0 + j;
        if (ti >= 0 && ti < 4096)
            acc += x[((size_t)(b * 4096) + ti) * 64 + c] * dw[c * 7 + j];
    }
    sm[c] = acc;
    __syncthreads();
    float o = 0.0f;
#pragma unroll 8
    for (int cc = 0; cc < 64; ++cc) o += sm[cc] * pw[c * 64 + cc];
    o = o / 1.0000049999875f * g[c] + bb[c];
    o = gelu_exact(o);
    out[((size_t)(b * 2048) + to) * 64 + c] = o;
}

__global__ __launch_bounds__(128) void conv2_k(
    const float* __restrict__ x, const float* __restrict__ dw,
    const float* __restrict__ pw, const float* __restrict__ g,
    const float* __restrict__ bb, float* __restrict__ out)
{
    const int to = blockIdx.x;   // 0..1023
    const int b  = blockIdx.y;
    const int c  = threadIdx.x;  // 0..127
    __shared__ float sm[64];
    if (c < 64) {
        float acc = 0.0f;
        const int t0 = 2 * to - 3;
#pragma unroll
        for (int j = 0; j < 7; ++j) {
            int ti = t0 + j;
            if (ti >= 0 && ti < 2048)
                acc += x[((size_t)(b * 2048) + ti) * 64 + c] * dw[c * 7 + j];
        }
        sm[c] = acc;
    }
    __syncthreads();
    float o = 0.0f;
#pragma unroll 8
    for (int cc = 0; cc < 64; ++cc) o += sm[cc] * pw[c * 64 + cc];
    o = o / 1.0000049999875f * g[c] + bb[c];
    o = gelu_exact(o);
    out[((size_t)(b * 1024) + to) * 128 + c] = o;
}

// ---------------------------------------------------------------------------
// WMMA GEMM: C[M,N] = act(A[M,K] * W[N,K]^T + bias[N]), W pre-converted bf16.
// Block = 128 threads (4 waves); block tile 128(M) x 64(N); wave tile 32x64.
// Double-buffered LDS weight tiles (64x32 bf16 = 4KB each), DMA-staged with
// GLOBAL_LOAD_ASYNC_TO_LDS_B128; next tile's DMA overlaps current tile's
// WMMAs (s_wait_asynccnt 2 exploits in-order async completion). B-fragments
// are raw bf16 ds loads (no conversions in the inner loop); all 4 fragments
// are register-resident before the 8 back-to-back WMMAs.
// Fragment layouts per CDNA5 ISA 7.12.2 (wave32):
//   A 16x32 bf16: lane(mlo,hi): elem j -> k = (j<8 ? j : 16+j-8) + 8*hi, row=mlo
//   B 32x16 bf16: lane(n=mlo,hi): elem j -> k = (j<8 ? j : 16+j-8) + 8*hi         (B[k][n]=W[n][k])
//   C/D 16x16 f32: lane: col n = mlo; VGPR r -> row = r + 8*hi
// ---------------------------------------------------------------------------
__global__ __launch_bounds__(128) void gemm_bias_act(
    const float* __restrict__ A, const __bf16* __restrict__ W,
    const float* __restrict__ bias, float* __restrict__ C,
    int M, int N, int K, int act)
{
    __shared__ __align__(16) __bf16 smW[2][64 * 32];   // 2 x 4KB

    const int tid  = threadIdx.x;
    const int lane = tid & 31;
    const int wave = tid >> 5;
    const int mlo  = lane & 15;
    const int hi   = lane >> 4;
    const int mBase = blockIdx.x * 128 + wave * 32;
    const int nBase = blockIdx.y * 64;

    int ar0 = mBase + mlo;       if (ar0 > M - 1) ar0 = M - 1;
    int ar1 = mBase + 16 + mlo;  if (ar1 > M - 1) ar1 = M - 1;
    const float* Ap0 = A + (size_t)ar0 * K;
    const float* Ap1 = A + (size_t)ar1 * K;

    // staging map: 128 threads x 2 issues x 16B covers 64 rows x 32 bf16
    const int srow = tid >> 2;        // 0..31
    const int scol = (tid & 3) * 8;   // bf16 index: 0,8,16,24

    const __bf16* Wb = W + (size_t)nBase * K;

    v8f acc[2][4];
#pragma unroll
    for (int mi = 0; mi < 2; ++mi)
#pragma unroll
        for (int nt = 0; nt < 4; ++nt) acc[mi][nt] = vzero8();

    // prologue: DMA tile k=0 into buffer 0
#pragma unroll
    for (int it = 0; it < 2; ++it) {
        const int r = srow + it * 32;
        async_g2l_b128(lds_off32(&smW[0][r * 32 + scol]),
                       Wb + (size_t)r * K + scol);
    }

    const int nk = K >> 5;
    for (int ks = 0; ks < nk; ++ks) {
        const int k0  = ks << 5;
        const int cur = ks & 1;
        const bool more = (ks + 1 < nk);

        // ---- DMA-stage NEXT weight tile (overlaps this step's math) ----
        if (more) {
#pragma unroll
            for (int it = 0; it < 2; ++it) {
                const int r = srow + it * 32;
                async_g2l_b128(lds_off32(&smW[cur ^ 1][r * 32 + scol]),
                               Wb + (size_t)r * K + k0 + 32 + scol);
            }
        }

        // ---- A fragments (global fp32 -> bf16; unique per wave) ----
        __builtin_prefetch(Ap0 + k0 + 64, 0, 1);
        v16bf a0, a1;
        {
            const float* p0 = Ap0 + k0 + 8 * hi;
            const float* p1 = Ap1 + k0 + 8 * hi;
#pragma unroll
            for (int j = 0; j < 8; ++j) {
                a0[j]     = f2bf(p0[j]);
                a0[8 + j] = f2bf(p0[16 + j]);
                a1[j]     = f2bf(p1[j]);
                a1[8 + j] = f2bf(p1[16 + j]);
            }
        }

        if (more) wait_asynccnt2(); else wait_asynccnt0();
        __syncthreads();

        // ---- all 4 B fragments register-resident, then 8 WMMAs ----
        v16bf bfr[4];
#pragma unroll
        for (int nt = 0; nt < 4; ++nt) {
            const __bf16* p = &smW[cur][(nt * 16 + mlo) * 32 + 8 * hi];
#pragma unroll
            for (int j = 0; j < 16; ++j) bfr[nt][j] = p[j < 8 ? j : j + 8];
        }
#pragma unroll
        for (int nt = 0; nt < 4; ++nt) {
            acc[0][nt] = __builtin_amdgcn_wmma_f32_16x16x32_bf16(
                false, a0, false, bfr[nt], (short)0, acc[0][nt], false, false);
            acc[1][nt] = __builtin_amdgcn_wmma_f32_16x16x32_bf16(
                false, a1, false, bfr[nt], (short)0, acc[1][nt], false, false);
        }
        wait_dscnt0();        // LDS reads retired before buffer reuse
        __syncthreads();
    }

    // ---- epilogue: bias + activation + guarded store ----
#pragma unroll
    for (int mi = 0; mi < 2; ++mi) {
#pragma unroll
        for (int nt = 0; nt < 4; ++nt) {
            const int col = nBase + nt * 16 + mlo;
            const float bv = bias ? bias[col] : 0.0f;
#pragma unroll
            for (int r = 0; r < 8; ++r) {
                int row = mBase + mi * 16 + r + 8 * hi;
                if (row < M) {
                    float v = acc[mi][nt][r] + bv;
                    if (act == 1) v = gelu_exact(v);
                    C[(size_t)row * N + col] = v;
                }
            }
        }
    }
}

// ---------------------------------------------------------------------------
// Insert cls token: h[b,0,:] = cls; h[b,1+t,:] = proj[b*1024+t,:]
// ---------------------------------------------------------------------------
__global__ void assemble_h(const float* __restrict__ proj,
                           const float* __restrict__ cls,
                           float* __restrict__ H)
{
    int idx = blockIdx.x * 256 + threadIdx.x;   // 0 .. 8*1025*256-1 (exact grid)
    int c = idx & 255;
    int r = (idx >> 8) % 1025;
    int b = idx / (1025 * 256);
    if (b >= 8) return;
    if (r == 0) H[idx] = cls[c];
    else        H[idx] = proj[((size_t)(b * 1024) + (r - 1)) * 256 + c];
}

// ---------------------------------------------------------------------------
// RoPE on q,k in place. [8,1025,8 heads,32] ; half = 16, rotate (d, d+16).
// ---------------------------------------------------------------------------
__global__ void rope_k(float* __restrict__ q, float* __restrict__ kk)
{
    int idx = blockIdx.x * 256 + threadIdx.x;
    const int total = 8 * 1025 * 8 * 16;
    if (idx >= total) return;
    int d    = idx & 15;
    int hh   = (idx >> 4) & 7;
    int rest = idx >> 7;
    int t    = rest % 1025;
    int b    = rest / 1025;
    float invf = __expf(-(float)d * (9.210340371976184f / 16.0f)); // 10000^(-d/16)
    float ang  = (float)t * invf;
    float sn, cs;
    sincosf(ang, &sn, &cs);
    size_t base = ((size_t)(b * 1025) + t) * 256 + hh * 32 + d;
    float x1 = q[base], x2 = q[base + 16];
    q[base]      = x1 * cs - x2 * sn;
    q[base + 16] = x1 * sn + x2 * cs;
    x1 = kk[base]; x2 = kk[base + 16];
    kk[base]      = x1 * cs - x2 * sn;
    kk[base + 16] = x1 * sn + x2 * cs;
}

// ---------------------------------------------------------------------------
// Flash attention, one wave per (b, h, 16-row q tile). hd = 32 (one WMMA K).
// Keys streamed in chunks of 32: V chunk is DMA-staged to LDS (async) while
// the 2 QK^T WMMAs + online softmax run; s_wait_asynccnt defers the wait to
// just before the 2 P*V WMMAs. P is restriped to A-fragment form through LDS.
// ---------------------------------------------------------------------------
__global__ __launch_bounds__(32) void flash_attn(
    const float* __restrict__ Q, const float* __restrict__ Kt,
    const float* __restrict__ V, float* __restrict__ O, int T)
{
    const int E    = 256;
    const int lane = threadIdx.x;
    const int mlo  = lane & 15;
    const int hi   = lane >> 4;
    const int q0   = blockIdx.x * 16;
    const int bh   = blockIdx.y;
    const int b    = bh >> 3, h = bh & 7;
    const size_t baseBT = (size_t)b * T;
    const float scale = 0.1767766952966369f;   // 1/sqrt(32)

    __shared__ float Pl[16][33];
    __shared__ __align__(16) float smV[32 * 32];   // V chunk [32 keys][32 dims]

    // Q A-fragment (rows q0..q0+15, full head dim 0..31)
    int qr = q0 + mlo;
    if (qr > T - 1) qr = T - 1;
    const float* qp = Q + (baseBT + qr) * E + h * 32;
    v16bf aq;
#pragma unroll
    for (int j = 0; j < 8; ++j) {
        aq[j]     = f2bf(qp[j + 8 * hi]);
        aq[8 + j] = f2bf(qp[16 + j + 8 * hi]);
    }

    v8f ctx0 = vzero8(), ctx1 = vzero8();
    float mrow[8], lsum[8];
#pragma unroll
    for (int r = 0; r < 8; ++r) { mrow[r] = -1e30f; lsum[r] = 0.0f; }

    const float* kb = Kt + baseBT * E + h * 32;
    const float* vb = V  + baseBT * E + h * 32;

    const int nch = (T + 31) / 32;
    for (int cc = 0; cc < nch; ++cc) {
        const int c0 = cc * 32;

        // ---- kick off async DMA of V chunk into LDS (overlaps softmax) ----
#pragma unroll
        for (int it = 0; it < 8; ++it) {
            int r  = it * 4 + (lane >> 3);            // 0..31
            int kk = c0 + r; if (kk > T - 1) kk = T - 1;   // P==0 there
            const float* gp = vb + (size_t)kk * E + (lane & 7) * 4;
            async_g2l_b128(lds_off32(&smV[r * 32 + (lane & 7) * 4]), gp);
        }

        // ---- K B-fragments: B[k][n] = K[key n][dim k] ----
        int key0 = c0 + mlo;
        int key1 = c0 + 16 + mlo;
        int kc0 = key0 > T - 1 ? T - 1 : key0;
        int kc1 = key1 > T - 1 ? T - 1 : key1;
        const float* p0 = kb + (size_t)kc0 * E + 8 * hi;
        const float* p1 = kb + (size_t)kc1 * E + 8 * hi;
        v16bf bk0, bk1;
#pragma unroll
        for (int j = 0; j < 16; ++j) { int jj = j < 8 ? j : j + 8; bk0[j] = f2bf(p0[jj]); bk1[j] = f2bf(p1[jj]); }
        v8f s0 = __builtin_amdgcn_wmma_f32_16x16x32_bf16(
            false, aq, false, bk0, (short)0, vzero8(), false, false);
        v8f s1 = __builtin_amdgcn_wmma_f32_16x16x32_bf16(
            false, aq, false, bk1, (short)0, vzero8(), false, false);
        const bool ok0 = key0 < T, ok1 = key1 < T;

        // ---- online softmax (rows r+8*hi live across 16-lane half) ----
#pragma unroll
        for (int r = 0; r < 8; ++r) {
            float f0 = ok0 ? s0[r] * scale : -1e30f;
            float f1 = ok1 ? s1[r] * scale : -1e30f;
            float cm = fmaxf(f0, f1);
            cm = fmaxf(cm, __shfl_xor(cm, 1, 32));
            cm = fmaxf(cm, __shfl_xor(cm, 2, 32));
            cm = fmaxf(cm, __shfl_xor(cm, 4, 32));
            cm = fmaxf(cm, __shfl_xor(cm, 8, 32));
            float mn   = fmaxf(mrow[r], cm);
            float corr = __expf(mrow[r] - mn);
            mrow[r] = mn;
            float pe0 = ok0 ? __expf(f0 - mn) : 0.0f;
            float pe1 = ok1 ? __expf(f1 - mn) : 0.0f;
            float cs = pe0 + pe1;
            cs += __shfl_xor(cs, 1, 32);
            cs += __shfl_xor(cs, 2, 32);
            cs += __shfl_xor(cs, 4, 32);
            cs += __shfl_xor(cs, 8, 32);
            lsum[r] = lsum[r] * corr + cs;
            ctx0[r] *= corr;
            ctx1[r] *= corr;
            Pl[r + 8 * hi][mlo]      = pe0;
            Pl[r + 8 * hi][16 + mlo] = pe1;
        }
        __syncthreads();
        // ---- restripe P (16x32) into an A fragment ----
        v16bf ap;
#pragma unroll
        for (int j = 0; j < 8; ++j) {
            ap[j]     = f2bf(Pl[mlo][j + 8 * hi]);
            ap[8 + j] = f2bf(Pl[mlo][16 + j + 8 * hi]);
        }

        // ---- V B-fragments from the DMA-staged LDS chunk ----
        wait_asynccnt0();
        __syncthreads();
        v16bf bv0, bv1;
#pragma unroll
        for (int j = 0; j < 16; ++j) {
            const float* pv = &smV[((j < 8 ? j : j + 8) + 8 * hi) * 32];
            bv0[j] = f2bf(pv[mlo]);
            bv1[j] = f2bf(pv[16 + mlo]);
        }
        ctx0 = __builtin_amdgcn_wmma_f32_16x16x32_bf16(
            false, ap, false, bv0, (short)0, ctx0, false, false);
        ctx1 = __builtin_amdgcn_wmma_f32_16x16x32_bf16(
            false, ap, false, bv1, (short)0, ctx1, false, false);

        wait_dscnt0();       // LDS reads retired before next chunk's DMA
        __syncthreads();
    }

#pragma unroll
    for (int r = 0; r < 8; ++r) {
        int row = q0 + r + 8 * hi;
        if (row < T) {
            float inv = 1.0f / lsum[r];
            float* op = O + (baseBT + row) * E + h * 32;
            op[mlo]      = ctx0[r] * inv;
            op[16 + mlo] = ctx1[r] * inv;
        }
    }
}

// ---------------------------------------------------------------------------
// LayerNorm over E=256 (one wave per row), optional fused residual add.
// ---------------------------------------------------------------------------
__global__ __launch_bounds__(32) void ln_kernel(
    const float* __restrict__ X, const float* __restrict__ R,
    const float* __restrict__ g, const float* __restrict__ bta,
    float* __restrict__ Y)
{
    const int row  = blockIdx.x;
    const int lane = threadIdx.x;
    const float* xp = X + (size_t)row * 256;
    const float* rp = R ? R + (size_t)row * 256 : nullptr;
    float vals[8];
    float s = 0.0f, sq = 0.0f;
#pragma unroll
    for (int j = 0; j < 8; ++j) {
        int i = j * 32 + lane;
        float v = xp[i];
        if (rp) v += rp[i];
        vals[j] = v;
        s += v;
        sq += v * v;
    }
    for (int m = 1; m < 32; m <<= 1) {
        s  += __shfl_xor(s,  m, 32);
        sq += __shfl_xor(sq, m, 32);
    }
    float mean = s * (1.0f / 256.0f);
    float var  = sq * (1.0f / 256.0f) - mean * mean;
    float inv  = rsqrtf(var + 1e-5f);
    float* yp = Y + (size_t)row * 256;
#pragma unroll
    for (int j = 0; j < 8; ++j) {
        int i = j * 32 + lane;
        yp[i] = (vals[j] - mean) * inv * g[i] + bta[i];
    }
}

// ---------------------------------------------------------------------------
// logits[b, t] = dot(h[b, 1+t, :], ts_w) + ts_b  (one wave per output)
// ---------------------------------------------------------------------------
__global__ __launch_bounds__(32) void token_scores(
    const float* __restrict__ H, const float* __restrict__ w,
    const float* __restrict__ bb, float* __restrict__ out)
{
    int row = blockIdx.x;               // 0..8191
    int b = row >> 10, t = (row & 1023) + 1;
    const float* hp = H + ((size_t)(b * 1025) + t) * 256;
    int lane = threadIdx.x;
    float s = 0.0f;
#pragma unroll
    for (int j = 0; j < 8; ++j) {
        int i = j * 32 + lane;
        s += hp[i] * w[i];
    }
    for (int m = 1; m < 32; m <<= 1) s += __shfl_xor(s, m, 32);
    if (lane == 0) out[row] = s + bb[0];
}

// ---------------------------------------------------------------------------
// Host-side orchestration
// ---------------------------------------------------------------------------
extern "C" void kernel_launch(void* const* d_in, const int* in_sizes, int n_in,
                              void* d_out, int out_size, void* d_ws, size_t ws_size,
                              hipStream_t stream)
{
    (void)in_sizes; (void)n_in; (void)out_size; (void)ws_size;
    const float* x        = (const float*)d_in[0];
    const float* conv1_dw = (const float*)d_in[1];
    const float* conv1_pw = (const float*)d_in[2];
    const float* bn1_g    = (const float*)d_in[3];
    const float* bn1_b    = (const float*)d_in[4];
    const float* conv2_dw = (const float*)d_in[5];
    const float* conv2_pw = (const float*)d_in[6];
    const float* bn2_g    = (const float*)d_in[7];
    const float* bn2_b    = (const float*)d_in[8];
    const float* proj_w   = (const float*)d_in[9];
    const float* proj_b   = (const float*)d_in[10];
    const float* cls_tok  = (const float*)d_in[11];
    const float* Wq       = (const float*)d_in[12];
    const float* bq       = (const float*)d_in[13];
    const float* Wk       = (const float*)d_in[14];
    const float* bk       = (const float*)d_in[15];
    const float* Wv       = (const float*)d_in[16];
    const float* bv       = (const float*)d_in[17];
    const float* Wo       = (const float*)d_in[18];
    const float* bo       = (const float*)d_in[19];
    const float* ln1_g    = (const float*)d_in[20];
    const float* ln1_b    = (const float*)d_in[21];
    const float* mlp_w1   = (const float*)d_in[22];
    const float* mlp_b1   = (const float*)d_in[23];
    const float* mlp_w2   = (const float*)d_in[24];
    const float* mlp_b2   = (const float*)d_in[25];
    const float* ln2_g    = (const float*)d_in[26];
    const float* ln2_b    = (const float*)d_in[27];
    const float* lat_w    = (const float*)d_in[28];
    const float* lat_b    = (const float*)d_in[29];
    const float* latn_g   = (const float*)d_in[30];
    const float* latn_b   = (const float*)d_in[31];
    const float* ts_w     = (const float*)d_in[32];
    const float* ts_b     = (const float*)d_in[33];

    float* ws = (float*)d_ws;
    size_t off = 0;
    float* y1  = ws + off; off += (size_t)8 * 2048 * 64;
    float* y2  = ws + off; off += (size_t)8 * 1024 * 128;
    float* h   = ws + off; off += (size_t)8 * 1025 * 256;
    float* qb  = ws + off; off += (size_t)8 * 1025 * 256;
    float* kb  = ws + off; off += (size_t)8 * 1025 * 256;
    float* vb  = ws + off; off += (size_t)8 * 1025 * 256;
    float* ctx = ws + off; off += (size_t)8 * 1025 * 256;
    float* tmp = ws + off; off += (size_t)8 * 1025 * 256;
    float* m1  = ws + off; off += (size_t)8200 * 1024;

    // bf16 weight mirrors (16B-aligned: 'off' is a multiple of 16 floats)
    __bf16* wsb = (__bf16*)(ws + off);
    size_t ob = 0;
    __bf16* projw_b = wsb + ob; ob += (size_t)256 * 128;
    __bf16* latw_b  = wsb + ob; ob += (size_t)256 * 256;
    __bf16* Wq_b    = wsb + ob; ob += (size_t)6 * 256 * 256;
    __bf16* Wk_b    = wsb + ob; ob += (size_t)6 * 256 * 256;
    __bf16* Wv_b    = wsb + ob; ob += (size_t)6 * 256 * 256;
    __bf16* Wo_b    = wsb + ob; ob += (size_t)6 * 256 * 256;
    __bf16* w1_b    = wsb + ob; ob += (size_t)6 * 1024 * 256;
    __bf16* w2_b    = wsb + ob; ob += (size_t)6 * 256 * 1024;

    // --- weight pre-conversion to bf16 (deterministic; once per call) ---
    cvt_w_bf16<<<dim3(128),  256, 0, stream>>>(proj_w, projw_b, 256 * 128);
    cvt_w_bf16<<<dim3(256),  256, 0, stream>>>(lat_w,  latw_b,  256 * 256);
    cvt_w_bf16<<<dim3(1536), 256, 0, stream>>>(Wq, Wq_b, 6 * 256 * 256);
    cvt_w_bf16<<<dim3(1536), 256, 0, stream>>>(Wk, Wk_b, 6 * 256 * 256);
    cvt_w_bf16<<<dim3(1536), 256, 0, stream>>>(Wv, Wv_b, 6 * 256 * 256);
    cvt_w_bf16<<<dim3(1536), 256, 0, stream>>>(Wo, Wo_b, 6 * 256 * 256);
    cvt_w_bf16<<<dim3(6144), 256, 0, stream>>>(mlp_w1, w1_b, 6 * 1024 * 256);
    cvt_w_bf16<<<dim3(6144), 256, 0, stream>>>(mlp_w2, w2_b, 6 * 256 * 1024);

    // --- conv frontend ---
    conv1_k<<<dim3(2048, 8), 64, 0, stream>>>(x, conv1_dw, conv1_pw, bn1_g, bn1_b, y1);
    conv2_k<<<dim3(1024, 8), 128, 0, stream>>>(y1, conv2_dw, conv2_pw, bn2_g, bn2_b, y2);

    // --- projection to E=256 and cls token ---
    gemm_bias_act<<<dim3(64, 4), 128, 0, stream>>>(y2, projw_b, proj_b, tmp,
                                                   8192, 256, 128, 0);
    assemble_h<<<dim3(8200), 256, 0, stream>>>(tmp, cls_tok, h);

    const int M = 8200, T = 1025;
    const int GX = 65;   // ceil(8200 / 128)
    for (int i = 0; i < 6; ++i) {
        const __bf16* Wqi = Wq_b + (size_t)i * 256 * 256;
        const __bf16* Wki = Wk_b + (size_t)i * 256 * 256;
        const __bf16* Wvi = Wv_b + (size_t)i * 256 * 256;
        const __bf16* Woi = Wo_b + (size_t)i * 256 * 256;
        const __bf16* w1i = w1_b + (size_t)i * 1024 * 256;
        const __bf16* w2i = w2_b + (size_t)i * 256 * 1024;

        gemm_bias_act<<<dim3(GX, 4), 128, 0, stream>>>(h, Wqi, bq + i * 256, qb, M, 256, 256, 0);
        gemm_bias_act<<<dim3(GX, 4), 128, 0, stream>>>(h, Wki, bk + i * 256, kb, M, 256, 256, 0);
        gemm_bias_act<<<dim3(GX, 4), 128, 0, stream>>>(h, Wvi, bv + i * 256, vb, M, 256, 256, 0);
        rope_k<<<dim3(4100), 256, 0, stream>>>(qb, kb);
        flash_attn<<<dim3(65, 64), 32, 0, stream>>>(qb, kb, vb, ctx, T);
        gemm_bias_act<<<dim3(GX, 4), 128, 0, stream>>>(ctx, Woi, bo + i * 256, tmp, M, 256, 256, 0);
        ln_kernel<<<dim3(M), 32, 0, stream>>>(h, tmp, ln1_g + i * 256, ln1_b + i * 256, h);
        gemm_bias_act<<<dim3(GX, 16), 128, 0, stream>>>(h, w1i, mlp_b1 + i * 1024, m1, M, 1024, 256, 1);
        gemm_bias_act<<<dim3(GX, 4), 128, 0, stream>>>(m1, w2i, mlp_b2 + i * 256, tmp, M, 256, 1024, 0);
        ln_kernel<<<dim3(M), 32, 0, stream>>>(h, tmp, ln2_g + i * 256, ln2_b + i * 256, h);
    }

    // --- final latent projection + LN + token scores ---
    gemm_bias_act<<<dim3(GX, 4), 128, 0, stream>>>(h, latw_b, lat_b, tmp, M, 256, 256, 0);
    ln_kernel<<<dim3(M), 32, 0, stream>>>(tmp, nullptr, latn_g, latn_b, h);
    token_scores<<<dim3(8192), 32, 0, stream>>>(h, ts_w, ts_b, (float*)d_out);
}
